// LSTM_48009144435064
// MI455X (gfx1250) — compile-verified, matrix operand order checked by reading, not run on hardware
//
#include <hip/hip_runtime.h>

#define SEQ     4096
#define IN_DIM  128
#define H_DIM   1024
#define G4H     4096   // 4*H
#define OUT_DIM 64

typedef __attribute__((ext_vector_type(16))) __bf16 v16bf;
typedef __attribute__((ext_vector_type(8)))  __bf16 v8bf;
typedef __attribute__((ext_vector_type(8)))  float  v8f;

// ---------------------------------------------------------------------------
// f32 -> bf16 split (hi + lo) for near-fp32 WMMA GEMMs
// ---------------------------------------------------------------------------
__global__ __launch_bounds__(256) void split_bf16_kernel(
    const float* __restrict__ src, __bf16* __restrict__ hi,
    __bf16* __restrict__ lo, int n) {
  int i = blockIdx.x * 256 + threadIdx.x;
  if (i < n) {
    float x = src[i];
    __bf16 h = (__bf16)x;
    hi[i] = h;
    lo[i] = (__bf16)(x - (float)h);
  }
}

// ---------------------------------------------------------------------------
// WMMA GEMM: C[m][n] = sum_k A[m][k]*B[n][k] + bias0[n] + bias1[n]
// A: M x K row-major (hi/lo bf16), B: N x K row-major (weights, hi/lo bf16)
// Split-precision: hi*hi + hi*lo + lo*hi  (3x v_wmma_f32_16x16x32_bf16)
// Each wave register-blocks a 32x32 output tile (2x2 WMMA tiles):
//   8 fragment loads -> 12 WMMAs per K=32 slab (2x intensity of 1-tile/wave).
// Block = 8 waves arranged 4(M) x 2(N) => 128x64 macro-tile.
// Per ISA 7.12.2: 16-bit A 16x32 frag: lane(row = lane&15, khalf = lane>>4)
// holds K in [khalf*8, khalf*8+8) and [16+khalf*8, 16+khalf*8+8).
// ---------------------------------------------------------------------------
__device__ inline v16bf load_frag16(const __bf16* __restrict__ base, int ld,
                                    int row, int kk, int khalf) {
  const __bf16* p = base + (size_t)row * ld + kk + khalf * 8;
  v8bf a0 = *(const v8bf*)p;         // 16B aligned: ld,kk mult of 32|128
  v8bf a1 = *(const v8bf*)(p + 16);
  v16bf v;
#pragma unroll
  for (int i = 0; i < 8; i++) { v[i] = a0[i]; v[i + 8] = a1[i]; }
  return v;
}

#define WMMA_BF16(A, B, C) \
  __builtin_amdgcn_wmma_f32_16x16x32_bf16(false, (A), false, (B), (short)0, (C), false, false)

__device__ inline void store_tile(float* __restrict__ C, int N, const v8f& c,
                                  int mt, int nt, int lane,
                                  const float* __restrict__ bias0,
                                  const float* __restrict__ bias1) {
  // C/D layout: lane -> col = lane&15, rows = vgpr r + 8*(lane>>4)
  const int cn = nt + (lane & 15);
  const float b = bias0[cn] + bias1[cn];
  const int mbase = mt + 8 * (lane >> 4);
#pragma unroll
  for (int r = 0; r < 8; r++) C[(size_t)(mbase + r) * N + cn] = c[r] + b;
}

__global__ __launch_bounds__(256) void wmma_gemm_bias_kernel(
    const __bf16* __restrict__ Ahi, const __bf16* __restrict__ Alo,
    const __bf16* __restrict__ Bhi, const __bf16* __restrict__ Blo,
    const float* __restrict__ bias0, const float* __restrict__ bias1,
    float* __restrict__ C, int M, int N, int K) {
  const int wave  = threadIdx.x >> 5;
  const int lane  = threadIdx.x & 31;
  const int wm    = wave & 3;   // 4 waves along M
  const int wn    = wave >> 2;  // 2 waves along N
  const int m0    = blockIdx.y * 128 + wm * 32;  // wave's two 16-row tiles
  const int n0    = blockIdx.x * 64 + wn * 32;   // wave's two 16-col tiles
  const int r     = lane & 15;
  const int khalf = lane >> 4;

  v8f c00 = {}, c01 = {}, c10 = {}, c11 = {};
  for (int kk = 0; kk < K; kk += 32) {
    v16bf a0h = load_frag16(Ahi, K, m0 + r,      kk, khalf);
    v16bf a0l = load_frag16(Alo, K, m0 + r,      kk, khalf);
    v16bf a1h = load_frag16(Ahi, K, m0 + 16 + r, kk, khalf);
    v16bf a1l = load_frag16(Alo, K, m0 + 16 + r, kk, khalf);
    v16bf b0h = load_frag16(Bhi, K, n0 + r,      kk, khalf);
    v16bf b0l = load_frag16(Blo, K, n0 + r,      kk, khalf);
    v16bf b1h = load_frag16(Bhi, K, n0 + 16 + r, kk, khalf);
    v16bf b1l = load_frag16(Blo, K, n0 + 16 + r, kk, khalf);

    c00 = WMMA_BF16(a0h, b0h, c00);
    c00 = WMMA_BF16(a0h, b0l, c00);
    c00 = WMMA_BF16(a0l, b0h, c00);

    c01 = WMMA_BF16(a0h, b1h, c01);
    c01 = WMMA_BF16(a0h, b1l, c01);
    c01 = WMMA_BF16(a0l, b1h, c01);

    c10 = WMMA_BF16(a1h, b0h, c10);
    c10 = WMMA_BF16(a1h, b0l, c10);
    c10 = WMMA_BF16(a1l, b0h, c10);

    c11 = WMMA_BF16(a1h, b1h, c11);
    c11 = WMMA_BF16(a1h, b1l, c11);
    c11 = WMMA_BF16(a1l, b1h, c11);
  }
  store_tile(C, N, c00, m0,      n0,      lane, bias0, bias1);
  store_tile(C, N, c01, m0,      n0 + 16, lane, bias0, bias1);
  store_tile(C, N, c10, m0 + 16, n0,      lane, bias0, bias1);
  store_tile(C, N, c11, m0 + 16, n0 + 16, lane, bias0, bias1);
}

// ---------------------------------------------------------------------------
// Persistent LSTM recurrence. 64 blocks x 256 threads. Each block owns 16
// hidden units => 64 W_hh rows (4 gates x 16 units) = 256KB resident in LDS
// (CDNA5: 320KB/WGP). Per step: broadcast h (4KB) to LDS, 8 LDS-dot-products
// per wave, gate math on lanes 0/1, ping-pong h in global, L2-atomic grid
// barrier between steps.
// ---------------------------------------------------------------------------
__global__ __launch_bounds__(256) void lstm_recur_kernel(
    const float* __restrict__ gx,     // T x 4H  (precomputed input gates + biases)
    const float* __restrict__ Whh,    // 4H x H
    const float* __restrict__ h_init, // H
    const float* __restrict__ c_init, // H
    float* __restrict__ hping,        // 2 x H ping-pong
    float* __restrict__ hs_out,       // T x H (or nullptr)
    unsigned* __restrict__ sync, int T) {
  extern __shared__ float smem[];
  float* Ws = smem;                 // 64 rows x 1024
  float* hs = smem + 64 * H_DIM;    // 1024
  const int tid  = threadIdx.x;
  const int wave = tid >> 5;
  const int lane = tid & 31;
  const int j0   = blockIdx.x * 16;       // first hidden unit of this block
  const unsigned nb = gridDim.x;

  // Stage this block's W_hh slice into LDS: row r = unit(r>>2), gate(r&3)
  for (int idx = tid; idx < 64 * (H_DIM / 4); idx += 256) {
    int r  = idx >> 8;          // /256 float4 per row
    int kq = idx & 255;
    int u = r >> 2, g = r & 3;
    ((float4*)Ws)[idx] =
        ((const float4*)(Whh + (size_t)(g * H_DIM + j0 + u) * H_DIM))[kq];
  }
  float c_reg = 0.f;
  if (lane < 2) c_reg = c_init[j0 + wave * 2 + lane];
  __syncthreads();

  for (int t = 0; t < T; ++t) {
    const float* hsrc = (t == 0) ? h_init : (hping + (size_t)(t & 1) * H_DIM);
    ((float4*)hs)[tid] = ((const float4*)hsrc)[tid];  // 1024 floats
    __syncthreads();

    float acc[8];
#pragma unroll
    for (int r = 0; r < 8; r++) acc[r] = 0.f;
    const float* wb = Ws + (size_t)(wave * 8) * H_DIM;  // this wave's 8 rows
    for (int k = lane; k < H_DIM; k += 32) {
      float hv = hs[k];
#pragma unroll
      for (int r = 0; r < 8; r++) acc[r] = fmaf(wb[r * H_DIM + k], hv, acc[r]);
    }
#pragma unroll
    for (int off = 16; off > 0; off >>= 1) {
#pragma unroll
      for (int r = 0; r < 8; r++) acc[r] += __shfl_xor(acc[r], off, 32);
    }
    if (lane < 2) {  // lane u owns unit wave*2+u; rows acc[u*4 .. u*4+3]
      const int j = j0 + wave * 2 + lane;
      const float* gxt = gx + (size_t)t * G4H;
      float pi = gxt[j]             + acc[lane * 4 + 0];
      float pf = gxt[H_DIM + j]     + acc[lane * 4 + 1];
      float pg = gxt[2 * H_DIM + j] + acc[lane * 4 + 2];
      float po = gxt[3 * H_DIM + j] + acc[lane * 4 + 3];
      float ig = 1.f / (1.f + __expf(-pi));
      float fg = 1.f / (1.f + __expf(-pf));
      float gg = tanhf(pg);
      float og = 1.f / (1.f + __expf(-po));
      c_reg = fg * c_reg + ig * gg;
      float hn = og * tanhf(c_reg);
      hping[(size_t)((t + 1) & 1) * H_DIM + j] = hn;
      if (hs_out) hs_out[(size_t)t * H_DIM + j] = hn;
    }
    // ---- grid barrier (release h stores, acquire others') ----
    __threadfence();
    __syncthreads();
    if (tid == 0) {
      __hip_atomic_fetch_add(sync, 1u, __ATOMIC_RELEASE, __HIP_MEMORY_SCOPE_AGENT);
      unsigned target = nb * (unsigned)(t + 1);
      while (__hip_atomic_load(sync, __ATOMIC_ACQUIRE, __HIP_MEMORY_SCOPE_AGENT) < target)
        __builtin_amdgcn_s_sleep(2);
    }
    __syncthreads();
    __threadfence();
  }
}

// ---------------------------------------------------------------------------
// Final 64-wide projection of the last hidden state (exact fp32)
// ---------------------------------------------------------------------------
__global__ __launch_bounds__(64) void final_linear_kernel(
    const float* __restrict__ h, const float* __restrict__ Wlin,
    const float* __restrict__ blin, float* __restrict__ out) {
  int o = threadIdx.x;
  if (o < OUT_DIM) {
    float s = blin[o];
    const float* w = Wlin + (size_t)o * H_DIM;
    for (int k = 0; k < H_DIM; k++) s = fmaf(w[k], h[k], s);
    out[o] = s;
  }
}

// ---------------------------------------------------------------------------
extern "C" void kernel_launch(void* const* d_in, const int* in_sizes, int n_in,
                              void* d_out, int out_size, void* d_ws, size_t ws_size,
                              hipStream_t stream) {
  (void)in_sizes; (void)n_in; (void)out_size; (void)ws_size;
  const float* x    = (const float*)d_in[0];
  const float* h0   = (const float*)d_in[1];   // (2,1,H)
  const float* c0   = (const float*)d_in[2];   // (2,1,H)
  const float* Wih0 = (const float*)d_in[3];
  const float* Whh0 = (const float*)d_in[4];
  const float* bih0 = (const float*)d_in[5];
  const float* bhh0 = (const float*)d_in[6];
  const float* Wih1 = (const float*)d_in[7];
  const float* Whh1 = (const float*)d_in[8];
  const float* bih1 = (const float*)d_in[9];
  const float* bhh1 = (const float*)d_in[10];
  const float* Wlin = (const float*)d_in[11];
  const float* blin = (const float*)d_in[12];

  // ---- workspace carve-up (256B aligned) ----
  char* base = (char*)d_ws;
  size_t off = 0;
  auto carve = [&](size_t bytes) -> char* {
    char* p = base + off;
    off = (off + bytes + 255) & ~(size_t)255;
    return p;
  };
  float*  gx    = (float*)carve((size_t)SEQ * G4H * 4);   // 64 MB (reused L0/L1)
  float*  hs0   = (float*)carve((size_t)SEQ * H_DIM * 4); // 16 MB
  __bf16* xhi   = (__bf16*)carve((size_t)SEQ * IN_DIM * 2);
  __bf16* xlo   = (__bf16*)carve((size_t)SEQ * IN_DIM * 2);
  __bf16* w0hi  = (__bf16*)carve((size_t)G4H * IN_DIM * 2);
  __bf16* w0lo  = (__bf16*)carve((size_t)G4H * IN_DIM * 2);
  __bf16* w1hi  = (__bf16*)carve((size_t)G4H * H_DIM * 2);
  __bf16* w1lo  = (__bf16*)carve((size_t)G4H * H_DIM * 2);
  __bf16* s0hi  = (__bf16*)carve((size_t)SEQ * H_DIM * 2);
  __bf16* s0lo  = (__bf16*)carve((size_t)SEQ * H_DIM * 2);
  float*  hping = (float*)carve(2 * H_DIM * 4);
  unsigned* sync0 = (unsigned*)carve(256);
  unsigned* sync1 = sync0 + 32;  // second counter in same 256B blob

  const int LDS_BYTES = (64 * H_DIM + H_DIM) * 4;  // 266,240 B (< 320KB/WGP)
  hipFuncSetAttribute((const void*)lstm_recur_kernel,
                      hipFuncAttributeMaxDynamicSharedMemorySize, LDS_BYTES);

  hipMemsetAsync(sync0, 0, 256, stream);

  // bf16 hi/lo staging of x, W_ih0, W_ih1
  auto nblk = [](int n) { return (n + 255) / 256; };
  split_bf16_kernel<<<nblk(SEQ * IN_DIM), 256, 0, stream>>>(x, xhi, xlo, SEQ * IN_DIM);
  split_bf16_kernel<<<nblk(G4H * IN_DIM), 256, 0, stream>>>(Wih0, w0hi, w0lo, G4H * IN_DIM);
  split_bf16_kernel<<<nblk(G4H * H_DIM), 256, 0, stream>>>(Wih1, w1hi, w1lo, G4H * H_DIM);

  // gx0 = x @ W_ih0^T + (b_ih0 + b_hh0)     [WMMA, split-precision bf16]
  wmma_gemm_bias_kernel<<<dim3(G4H / 64, SEQ / 128), 256, 0, stream>>>(
      xhi, xlo, w0hi, w0lo, bih0, bhh0, gx, SEQ, G4H, IN_DIM);

  // layer-0 recurrence (persistent, W_hh0 LDS-resident)
  lstm_recur_kernel<<<64, 256, LDS_BYTES, stream>>>(
      gx, Whh0, h0, c0, hping, hs0, sync0, SEQ);

  // gx1 = hs0 @ W_ih1^T + (b_ih1 + b_hh1)   [WMMA]
  split_bf16_kernel<<<nblk(SEQ * H_DIM), 256, 0, stream>>>(hs0, s0hi, s0lo, SEQ * H_DIM);
  wmma_gemm_bias_kernel<<<dim3(G4H / 64, SEQ / 128), 256, 0, stream>>>(
      s0hi, s0lo, w1hi, w1lo, bih1, bhh1, gx, SEQ, G4H, H_DIM);

  // layer-1 recurrence (final h lands in hping[ (SEQ&1)==0 -> buffer 0 ])
  lstm_recur_kernel<<<64, 256, LDS_BYTES, stream>>>(
      gx, Whh1, h0 + H_DIM, c0 + H_DIM, hping, nullptr, sync1, SEQ);

  // pred = h_last @ W_lin^T + b_lin
  final_linear_kernel<<<1, 64, 0, stream>>>(hping, Wlin, blin, (float*)d_out);
}
